// PointNet2_14577119002853
// MI455X (gfx1250) — matrix operand order repaired and checked by reading, not compiled, not run on hardware
//
#include <hip/hip_runtime.h>
#include <hip/hip_bf16.h>
#include <math.h>

// ---------------------------------------------------------------------------
// PointNet++ segmentation forward pass for gfx1250 (MI455X).
// MLP GEMMs: v_wmma_f32_16x16x32_f16, weights pre-packed into B-fragment
// order (f16), layer-1 activations staged in LDS in A-fragment order.
// ---------------------------------------------------------------------------

typedef __attribute__((ext_vector_type(16))) _Float16 v16h;
typedef __attribute__((ext_vector_type(8)))  float    v8f;

#define BN_EPS 1e-5f

struct LP  { const float *W, *b, *g, *be, *rm, *rv; };  // host-side layer handle
struct LPB { const float *b, *g, *be, *rm, *rv; };      // bias + BN params

__device__ __forceinline__ v8f wmma_f16(v16h a, v16h b, v8f c) {
  // D = A(16x32 f16) * B(32x16 f16) + C(16x16 f32)
  return __builtin_amdgcn_wmma_f32_16x16x32_f16(
      /*neg_a=*/false, a, /*neg_b=*/false, b,
      /*c_mod=*/(short)0, c, /*reuse_a=*/false, /*reuse_b=*/false);
}

// A-fragment (16x32 f16) K-offset for half-element h (0..15), lane-half hi.
// Layout per CDNA5 ISA "16-bit A-Matrix 16x32".
__device__ __forceinline__ int a_koff(int h, int hi) {
  int v = h >> 1, p = h & 1;
  return ((v >> 2) << 4) + hi * 8 + ((v & 3) << 1) + p;
}

// ---------------------------------------------------------------------------
// 0. Weight repack: W (Cin x C, f32 row-major) -> packed f16 B-fragments.
//    Fragment (ch, nt): 32 lanes x 16 halfs; element h of lane =
//    W[(ch*32 + (h>>3)*16 + hi*8 + (h&7)) * C + nt*16 + col], zero-padded past Cin.
// ---------------------------------------------------------------------------
__global__ void repack_w_kernel(const float* __restrict__ W, int Cin, int C,
                                _Float16* __restrict__ Wp, int total) {
  int idx = blockIdx.x * blockDim.x + threadIdx.x;
  if (idx >= total) return;
  int lane = idx & 31;
  int fn = idx >> 5;
  int ntiles = C >> 4;
  int nt = fn % ntiles;
  int ch = fn / ntiles;
  int col = lane & 15, hi = lane >> 4;
  _Float16* o = Wp + (size_t)idx * 16;
  for (int h = 0; h < 16; ++h) {
    int kb = ch * 32 + ((h >> 3) << 4) + hi * 8 + (h & 7);
    o[h] = (_Float16)((kb < Cin) ? W[(size_t)kb * C + nt * 16 + col] : 0.f);
  }
}

// ---------------------------------------------------------------------------
// 1. Extract pos0 (B,N,3) and x0 (B,N,7) from cloud_data (B,9,N)
// ---------------------------------------------------------------------------
__global__ void prep_kernel(const float* __restrict__ cloud,
                            float* __restrict__ pos0, float* __restrict__ x0) {
  int idx = blockIdx.x * blockDim.x + threadIdx.x;
  if (idx >= 16 * 4096) return;
  int b = idx >> 12, n = idx & 4095;
  const float* c = cloud + (size_t)b * 9 * 4096;
  pos0[(size_t)idx * 3 + 0] = c[0 * 4096 + n];
  pos0[(size_t)idx * 3 + 1] = c[1 * 4096 + n];
  pos0[(size_t)idx * 3 + 2] = c[2 * 4096 + n];
#pragma unroll
  for (int j = 0; j < 7; ++j) x0[(size_t)idx * 7 + j] = c[(2 + j) * 4096 + n];
}

// ---------------------------------------------------------------------------
// 2. Farthest point sampling (matches jax scan semantics; first-index argmax)
// ---------------------------------------------------------------------------
__global__ __launch_bounds__(256) void fps_kernel(const float* __restrict__ pos,
                                                  int N, int M, int* __restrict__ out) {
  extern __shared__ char smem_raw[];
  float* md = (float*)smem_raw;   // N floats
  float* rv = md + N;             // 256 floats
  int*   ri = (int*)(rv + 256);   // 256 ints
  int b = blockIdx.x, t = threadIdx.x;
  const float* P = pos + (size_t)b * N * 3;
  for (int i = t; i < N; i += 256) md[i] = 1e30f;
  __syncthreads();
  int cur = 0;
  for (int s = 0; s < M; ++s) {
    if (t == 0) out[(size_t)b * M + s] = cur;
    float cx = P[cur * 3 + 0], cy = P[cur * 3 + 1], cz = P[cur * 3 + 2];
    float bv = -1.f; int bi = 0;
    for (int i = t; i < N; i += 256) {
      float dx = P[i * 3 + 0] - cx, dy = P[i * 3 + 1] - cy, dz = P[i * 3 + 2] - cz;
      float d = dx * dx + dy * dy + dz * dz;
      float m = fminf(md[i], d);
      md[i] = m;
      if (m > bv) { bv = m; bi = i; }
    }
    rv[t] = bv; ri[t] = bi;
    __syncthreads();
    for (int o = 128; o > 0; o >>= 1) {
      if (t < o) {
        float v2 = rv[t + o]; int j2 = ri[t + o];
        if (v2 > rv[t] || (v2 == rv[t] && j2 < ri[t])) { rv[t] = v2; ri[t] = j2; }
      }
      __syncthreads();
    }
    cur = ri[0];
    __syncthreads();
  }
}

// ---------------------------------------------------------------------------
// 3. Gather sampled center positions
// ---------------------------------------------------------------------------
__global__ void gather_pos_kernel(const float* __restrict__ pos, const int* __restrict__ sidx,
                                  float* __restrict__ cpos, int M, int N, int B) {
  int idx = blockIdx.x * blockDim.x + threadIdx.x;
  if (idx >= B * M) return;
  int b = idx / M;
  const float* p = pos + ((size_t)b * N + sidx[idx]) * 3;
  cpos[(size_t)idx * 3 + 0] = p[0];
  cpos[(size_t)idx * 3 + 1] = p[1];
  cpos[(size_t)idx * 3 + 2] = p[2];
}

// ---------------------------------------------------------------------------
// 4. k-nearest within radius (lexicographic (d2, idx) selection == stable top_k)
// ---------------------------------------------------------------------------
__global__ void knn_radius_kernel(const float* __restrict__ cpos, const float* __restrict__ pos,
                                  int M, int N, int K, float r2,
                                  int* __restrict__ nidx, unsigned char* __restrict__ valid, int B) {
  int idx = blockIdx.x * blockDim.x + threadIdx.x;
  if (idx >= B * M) return;
  int b = idx / M;
  const float* P = pos + (size_t)b * N * 3;
  float cx = cpos[(size_t)idx * 3 + 0];
  float cy = cpos[(size_t)idx * 3 + 1];
  float cz = cpos[(size_t)idx * 3 + 2];
  float lastd = -1.f; int lasti = -1;
  for (int kk = 0; kk < K; ++kk) {
    float bd = __builtin_inff(); int bi = 0;
    for (int i = 0; i < N; ++i) {
      float dx = P[i * 3 + 0] - cx, dy = P[i * 3 + 1] - cy, dz = P[i * 3 + 2] - cz;
      float d = dx * dx + dy * dy + dz * dz;
      bool after = (d > lastd) || (d == lastd && i > lasti);
      if (after && d < bd) { bd = d; bi = i; }
    }
    nidx[(size_t)idx * K + kk] = bi;
    valid[(size_t)idx * K + kk] = (bd <= r2) ? 1 : 0;
    lastd = bd; lasti = bi;
  }
}

// ---------------------------------------------------------------------------
// 5. Fused PointConv: gather -> concat(xj, rel) -> (Linear/ReLU/BN)x2 ->
//    masked max over K neighbors.  One wave per center; templated shapes.
//    Layer-1 output scattered to LDS in A-fragment order for layer 2.
// ---------------------------------------------------------------------------
template <int CX, int C1, int C2>
__global__ __launch_bounds__(32) void sa_conv_max_kernel(
    const float* __restrict__ xsrc,
    const float* __restrict__ psrc, int Nsrc,
    const float* __restrict__ cpos,
    const int* __restrict__ nidx,
    const unsigned char* __restrict__ valid,
    int M, int K,
    const _Float16* __restrict__ Wp1, LPB p1,
    const _Float16* __restrict__ Wp2, LPB p2,
    float* __restrict__ out) {
  constexpr int CIN  = CX + 3;
  constexpr int NCH1 = (CIN + 31) / 32;
  constexpr int NCH2 = C1 / 32;
  constexpr int NT1  = C1 / 16;
  constexpr int NT2  = C2 / 16;
  extern __shared__ char smem_raw[];
  _Float16* h1f = (_Float16*)smem_raw;       // (K/16)*NCH2 fragments, 512 halfs each
  const int bm = blockIdx.x;                 // b*M + m
  const int b  = bm / M;
  const int lane = threadIdx.x;
  const int col = lane & 15, hi = lane >> 4;
  const int nkt = K >> 4;

  float cp0 = 0.f, cp1 = 0.f, cp2 = 0.f;
  if (cpos) { cp0 = cpos[(size_t)bm * 3 + 0]; cp1 = cpos[(size_t)bm * 3 + 1]; cp2 = cpos[(size_t)bm * 3 + 2]; }

  // ---- layer 1 ----
  for (int t = 0; t < nkt; ++t) {
    const int row = t * 16 + col;
    const int nid = nidx ? nidx[(size_t)bm * K + row] : row;
    const float* xrow = xsrc + ((size_t)b * Nsrc + nid) * CX;
    const float* prow = psrc + ((size_t)b * Nsrc + nid) * 3;
    const float rel0 = prow[0] - cp0, rel1 = prow[1] - cp1, rel2 = prow[2] - cp2;
    v16h a[NCH1];
#pragma unroll
    for (int ch = 0; ch < NCH1; ++ch) {
#pragma unroll
      for (int h = 0; h < 16; ++h) {
        const int kk = ch * 32 + a_koff(h, hi);
        const int kc = (kk < CX) ? kk : 0;
        float xv = xrow[kc];                       // unconditional, clamped addr
        float f = (kk < CX) ? xv : 0.f;            // branch-free selects
        f = (kk == CX)     ? rel0 : f;
        f = (kk == CX + 1) ? rel1 : f;
        f = (kk == CX + 2) ? rel2 : f;
        a[ch][h] = (_Float16)f;
      }
    }
#pragma unroll
    for (int nt = 0; nt < NT1; ++nt) {
      v8f acc = {};
#pragma unroll
      for (int ch = 0; ch < NCH1; ++ch) {
        v16h bf = *(const v16h*)(Wp1 + (size_t)((ch * NT1 + nt) * 32 + lane) * 16);
        acc = wmma_f16(a[ch], bf, acc);
      }
      const int ncol = nt * 16 + col;
      const float bias = p1.b[ncol];
      const float scl  = p1.g[ncol] * rsqrtf(p1.rv[ncol] + BN_EPS);
      const float rmu  = p1.rm[ncol], bta = p1.be[ncol];
      // scatter into layer-2 A-fragment order
      const int ko  = ncol & 31, ch2 = ncol >> 5;
      const int lhi = (ko >> 3) & 1;
      const int hh  = ((ko >> 4) * 4 + ((ko & 7) >> 1)) * 2 + (ko & 1);
#pragma unroll
      for (int i = 0; i < 8; ++i) {
        float v = fmaxf(acc[i] + bias, 0.f);
        v = scl * (v - rmu) + bta;
        const int r = hi * 8 + i;                  // row within this 16-tile
        h1f[(size_t)((t * NCH2 + ch2) * 32 + lhi * 16 + r) * 16 + hh] = (_Float16)v;
      }
    }
  }
  __syncthreads();

  // ---- layer 2 + masked max over neighbors ----
#pragma unroll
  for (int nt = 0; nt < NT2; ++nt) {
    const int ncol = nt * 16 + col;
    const float bias = p2.b[ncol];
    const float scl  = p2.g[ncol] * rsqrtf(p2.rv[ncol] + BN_EPS);
    const float rmu  = p2.rm[ncol], bta = p2.be[ncol];
    float rmax = -__builtin_inff();
    for (int t = 0; t < nkt; ++t) {
      v8f acc = {};
#pragma unroll
      for (int ch = 0; ch < NCH2; ++ch) {
        v16h a2 = *(const v16h*)(h1f + (size_t)((t * NCH2 + ch) * 32 + lane) * 16);
        v16h bf = *(const v16h*)(Wp2 + (size_t)((ch * NT2 + nt) * 32 + lane) * 16);
        acc = wmma_f16(a2, bf, acc);
      }
      float m = -__builtin_inff();
#pragma unroll
      for (int i = 0; i < 8; ++i) {
        float v = fmaxf(acc[i] + bias, 0.f);
        v = scl * (v - rmu) + bta;
        const int rrow = t * 16 + hi * 8 + i;
        bool ok = valid ? (valid[(size_t)bm * K + rrow] != 0) : true;
        v = ok ? v : -__builtin_inff();
        m = fmaxf(m, v);
      }
      m = fmaxf(m, __shfl_xor(m, 16, 32));         // fold rows r and r+8 halves
      rmax = fmaxf(rmax, m);
    }
    if (lane < 16) out[(size_t)bm * C2 + ncol] = rmax;
  }
}

// ---------------------------------------------------------------------------
// 6. Generic 2-layer MLP (Linear/ReLU/BN x2) over dense rows; wave per 16 rows.
// ---------------------------------------------------------------------------
template <int CIN, int C1, int C2>
__global__ __launch_bounds__(32) void mlp2_rows_kernel(
    const float* __restrict__ X,
    const _Float16* __restrict__ Wp1, LPB p1,
    const _Float16* __restrict__ Wp2, LPB p2,
    float* __restrict__ out) {
  constexpr int NCH1 = (CIN + 31) / 32;
  constexpr int NCH2 = C1 / 32;
  constexpr int NT1  = C1 / 16;
  constexpr int NT2  = C2 / 16;
  extern __shared__ char smem_raw[];
  _Float16* h1f = (_Float16*)smem_raw;       // NCH2 fragments, 512 halfs each
  const int lane = threadIdx.x;
  const int col = lane & 15, hi = lane >> 4;
  const size_t rbase = (size_t)blockIdx.x * 16;
  const float* xrow = X + (rbase + col) * CIN;

  v16h a[NCH1];
#pragma unroll
  for (int ch = 0; ch < NCH1; ++ch) {
#pragma unroll
    for (int h = 0; h < 16; ++h) {
      const int kk = ch * 32 + a_koff(h, hi);
      const int kc = (kk < CIN) ? kk : 0;
      float xv = xrow[kc];
      a[ch][h] = (_Float16)((kk < CIN) ? xv : 0.f);
    }
  }
#pragma unroll
  for (int nt = 0; nt < NT1; ++nt) {
    v8f acc = {};
#pragma unroll
    for (int ch = 0; ch < NCH1; ++ch) {
      v16h bf = *(const v16h*)(Wp1 + (size_t)((ch * NT1 + nt) * 32 + lane) * 16);
      acc = wmma_f16(a[ch], bf, acc);
    }
    const int ncol = nt * 16 + col;
    const float bias = p1.b[ncol];
    const float scl  = p1.g[ncol] * rsqrtf(p1.rv[ncol] + BN_EPS);
    const float rmu  = p1.rm[ncol], bta = p1.be[ncol];
    const int ko  = ncol & 31, ch2 = ncol >> 5;
    const int lhi = (ko >> 3) & 1;
    const int hh  = ((ko >> 4) * 4 + ((ko & 7) >> 1)) * 2 + (ko & 1);
#pragma unroll
    for (int i = 0; i < 8; ++i) {
      float v = fmaxf(acc[i] + bias, 0.f);
      v = scl * (v - rmu) + bta;
      const int r = hi * 8 + i;
      h1f[(size_t)(ch2 * 32 + lhi * 16 + r) * 16 + hh] = (_Float16)v;
    }
  }
  __syncthreads();

#pragma unroll
  for (int nt = 0; nt < NT2; ++nt) {
    v8f acc = {};
#pragma unroll
    for (int ch = 0; ch < NCH2; ++ch) {
      v16h a2 = *(const v16h*)(h1f + (size_t)(ch * 32 + lane) * 16);
      v16h bf = *(const v16h*)(Wp2 + (size_t)((ch * NT2 + nt) * 32 + lane) * 16);
      acc = wmma_f16(a2, bf, acc);
    }
    const int ncol = nt * 16 + col;
    const float bias = p2.b[ncol];
    const float scl  = p2.g[ncol] * rsqrtf(p2.rv[ncol] + BN_EPS);
    const float rmu  = p2.rm[ncol], bta = p2.be[ncol];
#pragma unroll
    for (int i = 0; i < 8; ++i) {
      float v = fmaxf(acc[i] + bias, 0.f);
      v = scl * (v - rmu) + bta;
      out[(rbase + hi * 8 + i) * C2 + ncol] = v;
    }
  }
}

// ---------------------------------------------------------------------------
// 7. FP3 input: concat(broadcast global feature, skip x2)  (k=1 interp == copy)
// ---------------------------------------------------------------------------
__global__ void concat_global_kernel(const float* __restrict__ xg, const float* __restrict__ x2,
                                     float* __restrict__ Xc, int M, int B) {
  int idx = blockIdx.x * blockDim.x + threadIdx.x;
  if (idx >= B * M) return;
  int b = idx / M;
  float* o = Xc + (size_t)idx * 192;
  const float* g = xg + (size_t)b * 128;
  for (int j = 0; j < 128; ++j) o[j] = g[j];
  const float* s = x2 + (size_t)idx * 64;
  for (int j = 0; j < 64; ++j) o[128 + j] = s[j];
}

// ---------------------------------------------------------------------------
// 8. 3-NN inverse-d2 interpolation + concat skip features
// ---------------------------------------------------------------------------
__global__ void interp3_concat_kernel(const float* __restrict__ xsrc, int Cs,
                                      const float* __restrict__ psrc, int Ns,
                                      const float* __restrict__ pdst, int Nd,
                                      const float* __restrict__ skip, int Csk,
                                      float* __restrict__ out, int B) {
  int idx = blockIdx.x * blockDim.x + threadIdx.x;
  if (idx >= B * Nd) return;
  int b = idx / Nd;
  const float* P = psrc + (size_t)b * Ns * 3;
  float px = pdst[(size_t)idx * 3 + 0];
  float py = pdst[(size_t)idx * 3 + 1];
  float pz = pdst[(size_t)idx * 3 + 2];
  float bd0 = __builtin_inff(), bd1 = __builtin_inff(), bd2 = __builtin_inff();
  int i0 = 0, i1 = 0, i2 = 0;
  for (int j = 0; j < Ns; ++j) {
    __builtin_prefetch(P + (j + 16) * 3, 0, 1);   // speculative; OOB dropped
    float dx = P[j * 3 + 0] - px, dy = P[j * 3 + 1] - py, dz = P[j * 3 + 2] - pz;
    float d = dx * dx + dy * dy + dz * dz;
    if (d < bd0)      { bd2 = bd1; i2 = i1; bd1 = bd0; i1 = i0; bd0 = d; i0 = j; }
    else if (d < bd1) { bd2 = bd1; i2 = i1; bd1 = d; i1 = j; }
    else if (d < bd2) { bd2 = d; i2 = j; }
  }
  float w0 = 1.f / fmaxf(bd0, 1e-16f);
  float w1 = 1.f / fmaxf(bd1, 1e-16f);
  float w2 = 1.f / fmaxf(bd2, 1e-16f);
  float ws = w0 + w1 + w2;
  const float* s0 = xsrc + ((size_t)b * Ns + i0) * Cs;
  const float* s1 = xsrc + ((size_t)b * Ns + i1) * Cs;
  const float* s2 = xsrc + ((size_t)b * Ns + i2) * Cs;
  float* o = out + (size_t)idx * (Cs + Csk);
  for (int c = 0; c < Cs; ++c) o[c] = (w0 * s0[c] + w1 * s1[c] + w2 * s2[c]) / ws;
  const float* sk = skip + (size_t)idx * Csk;
  for (int c = 0; c < Csk; ++c) o[Cs + c] = sk[c];
}

// ---------------------------------------------------------------------------
// 9. Head: lin1+ReLU, lin2, softmax; emit scores then proba, (B,8,N) each.
// ---------------------------------------------------------------------------
__global__ void head_kernel(const float* __restrict__ xin,
                            const float* __restrict__ W1, const float* __restrict__ b1,
                            const float* __restrict__ W2, const float* __restrict__ b2,
                            float* __restrict__ out, int B, int N) {
  int idx = blockIdx.x * blockDim.x + threadIdx.x;
  if (idx >= B * N) return;
  int b = idx / N, n = idx % N;
  const float* x = xin + (size_t)idx * 32;
  float h[32];
  for (int j = 0; j < 32; ++j) {
    float s = b1[j];
    for (int i = 0; i < 32; ++i) s += x[i] * W1[i * 32 + j];
    h[j] = fmaxf(s, 0.f);
  }
  float sc[8], e[8];
  float mx = -__builtin_inff();
  for (int c = 0; c < 8; ++c) {
    float s = b2[c];
    for (int i = 0; i < 32; ++i) s += h[i] * W2[i * 8 + c];
    sc[c] = s; mx = fmaxf(mx, s);
  }
  float den = 0.f;
  for (int c = 0; c < 8; ++c) { e[c] = expf(sc[c] - mx); den += e[c]; }
  size_t half = (size_t)B * 8 * N;
  for (int c = 0; c < 8; ++c) {
    out[((size_t)b * 8 + c) * N + n] = sc[c];
    out[half + ((size_t)b * 8 + c) * N + n] = e[c] / den;
  }
}

// ---------------------------------------------------------------------------
// Host orchestration
// ---------------------------------------------------------------------------
extern "C" void kernel_launch(void* const* d_in, const int* in_sizes, int n_in,
                              void* d_out, int out_size, void* d_ws, size_t ws_size,
                              hipStream_t stream) {
  (void)in_sizes; (void)n_in; (void)out_size; (void)ws_size;
  constexpr int B = 16, N = 4096, M1 = 1024, K1 = 32, M2 = 256, K2 = 64;
  const float r1sq = 0.15f * 0.15f, r2sq = 0.30f * 0.30f;

  const float* cloud = (const float*)d_in[0];
  auto mk = [&](int base) -> LP {
    return LP{ (const float*)d_in[base + 0], (const float*)d_in[base + 1],
               (const float*)d_in[base + 2], (const float*)d_in[base + 3],
               (const float*)d_in[base + 4], (const float*)d_in[base + 5] };
  };
  // Leaves in setup_inputs() insertion order: per layer {W,b,g,be,rm,rv}.
  LP m1a = mk(1),  m1b = mk(7);
  LP m2a = mk(13), m2b = mk(19);
  LP m3a = mk(25), m3b = mk(31);
  LP f3a = mk(37), f3b = mk(43);
  LP f2a = mk(49), f2b = mk(55);
  LP f1a = mk(61), f1b = mk(67);
  const float* lin1W = (const float*)d_in[73];
  const float* lin1b = (const float*)d_in[74];
  const float* lin2W = (const float*)d_in[75];
  const float* lin2b = (const float*)d_in[76];
  auto bn = [](LP l) -> LPB { return LPB{l.b, l.g, l.be, l.rm, l.rv}; };

  char* ws = (char*)d_ws;
  size_t off = 0;
  auto alloc = [&](size_t bytes) -> void* {
    void* p = ws + off;
    off += (bytes + 255) & ~(size_t)255;
    return p;
  };
  float* pos0  = (float*)alloc((size_t)B * N * 3 * 4);
  float* x0    = (float*)alloc((size_t)B * N * 7 * 4);
  int*   sidx1 = (int*)  alloc((size_t)B * M1 * 4);
  float* pos1  = (float*)alloc((size_t)B * M1 * 3 * 4);
  int*   nidx1 = (int*)  alloc((size_t)B * M1 * K1 * 4);
  unsigned char* val1 = (unsigned char*)alloc((size_t)B * M1 * K1);
  float* x1    = (float*)alloc((size_t)B * M1 * 32 * 4);
  int*   sidx2 = (int*)  alloc((size_t)B * M2 * 4);
  float* pos2  = (float*)alloc((size_t)B * M2 * 3 * 4);
  int*   nidx2 = (int*)  alloc((size_t)B * M2 * K2 * 4);
  unsigned char* val2 = (unsigned char*)alloc((size_t)B * M2 * K2);
  float* x2    = (float*)alloc((size_t)B * M2 * 64 * 4);
  float* xg    = (float*)alloc((size_t)B * 128 * 4);
  float* Xc3   = (float*)alloc((size_t)B * M2 * 192 * 4);
  float* xf3   = (float*)alloc((size_t)B * M2 * 64 * 4);
  float* Xc2   = (float*)alloc((size_t)B * M1 * 96 * 4);
  float* xf2   = (float*)alloc((size_t)B * M1 * 32 * 4);
  float* Xc1   = (float*)alloc((size_t)B * N * 39 * 4);
  float* xf1   = (float*)alloc((size_t)B * N * 32 * 4);

  // ---- pack all MLP weights into f16 B-fragment order ----
  auto packW = [&](const float* W, int Cin, int C) -> const _Float16* {
    int nch = (Cin + 31) / 32;
    int total = nch * (C / 16) * 32;               // fragments * 32 lanes
    _Float16* Wp = (_Float16*)alloc((size_t)total * 16 * 2);
    repack_w_kernel<<<dim3((total + 255) / 256), dim3(256), 0, stream>>>(W, Cin, C, Wp, total);
    return Wp;
  };
  const _Float16* Wp_m1a = packW(m1a.W, 10, 32);
  const _Float16* Wp_m1b = packW(m1b.W, 32, 32);
  const _Float16* Wp_m2a = packW(m2a.W, 35, 64);
  const _Float16* Wp_m2b = packW(m2b.W, 64, 64);
  const _Float16* Wp_m3a = packW(m3a.W, 67, 64);
  const _Float16* Wp_m3b = packW(m3b.W, 64, 128);
  const _Float16* Wp_f3a = packW(f3a.W, 192, 64);
  const _Float16* Wp_f3b = packW(f3b.W, 64, 64);
  const _Float16* Wp_f2a = packW(f2a.W, 96, 64);
  const _Float16* Wp_f2b = packW(f2b.W, 64, 32);
  const _Float16* Wp_f1a = packW(f1a.W, 39, 32);
  const _Float16* Wp_f1b = packW(f1b.W, 32, 32);

  // ---- input prep ----
  prep_kernel<<<dim3((B * N + 255) / 256), dim3(256), 0, stream>>>(cloud, pos0, x0);

  // ---- SA level 1 ----
  fps_kernel<<<dim3(B), dim3(256), (size_t)(N + 512) * 4, stream>>>(pos0, N, M1, sidx1);
  gather_pos_kernel<<<dim3((B * M1 + 255) / 256), dim3(256), 0, stream>>>(pos0, sidx1, pos1, M1, N, B);
  knn_radius_kernel<<<dim3((B * M1 + 255) / 256), dim3(256), 0, stream>>>(
      pos1, pos0, M1, N, K1, r1sq, nidx1, val1, B);
  sa_conv_max_kernel<7, 32, 32><<<dim3(B * M1), dim3(32), (size_t)(K1 / 16) * 1 * 1024, stream>>>(
      x0, pos0, N, pos1, nidx1, val1, M1, K1, Wp_m1a, bn(m1a), Wp_m1b, bn(m1b), x1);

  // ---- SA level 2 ----
  fps_kernel<<<dim3(B), dim3(256), (size_t)(M1 + 512) * 4, stream>>>(pos1, M1, M2, sidx2);
  gather_pos_kernel<<<dim3((B * M2 + 255) / 256), dim3(256), 0, stream>>>(pos1, sidx2, pos2, M2, M1, B);
  knn_radius_kernel<<<dim3((B * M2 + 255) / 256), dim3(256), 0, stream>>>(
      pos2, pos1, M2, M1, K2, r2sq, nidx2, val2, B);
  sa_conv_max_kernel<32, 64, 64><<<dim3(B * M2), dim3(32), (size_t)(K2 / 16) * 2 * 1024, stream>>>(
      x1, pos1, M1, pos2, nidx2, val2, M2, K2, Wp_m2a, bn(m2a), Wp_m2b, bn(m2b), x2);

  // ---- global SA (identity neighbors, rel = absolute pos, no mask) ----
  sa_conv_max_kernel<64, 64, 128><<<dim3(B), dim3(32), (size_t)(M2 / 16) * 2 * 1024, stream>>>(
      x2, pos2, M2, nullptr, nullptr, nullptr, 1, M2, Wp_m3a, bn(m3a), Wp_m3b, bn(m3b), xg);

  // ---- FP3: broadcast global feature + x2, MLP 192->64->64 ----
  concat_global_kernel<<<dim3((B * M2 + 255) / 256), dim3(256), 0, stream>>>(xg, x2, Xc3, M2, B);
  mlp2_rows_kernel<192, 64, 64><<<dim3(B * M2 / 16), dim3(32), 2 * 1024, stream>>>(
      Xc3, Wp_f3a, bn(f3a), Wp_f3b, bn(f3b), xf3);

  // ---- FP2: 3-NN interp pos2->pos1, concat x1, MLP 96->64->32 ----
  interp3_concat_kernel<<<dim3((B * M1 + 255) / 256), dim3(256), 0, stream>>>(
      xf3, 64, pos2, M2, pos1, M1, x1, 32, Xc2, B);
  mlp2_rows_kernel<96, 64, 32><<<dim3(B * M1 / 16), dim3(32), 2 * 1024, stream>>>(
      Xc2, Wp_f2a, bn(f2a), Wp_f2b, bn(f2b), xf2);

  // ---- FP1: 3-NN interp pos1->pos0, concat x0, MLP 39->32->32 ----
  interp3_concat_kernel<<<dim3((B * N + 255) / 256), dim3(256), 0, stream>>>(
      xf2, 32, pos1, M1, pos0, N, x0, 7, Xc1, B);
  mlp2_rows_kernel<39, 32, 32><<<dim3(B * N / 16), dim3(32), 1 * 1024, stream>>>(
      Xc1, Wp_f1a, bn(f1a), Wp_f1b, bn(f1b), xf1);

  // ---- classifier head + softmax, transposed output ----
  head_kernel<<<dim3((B * N + 255) / 256), dim3(256), 0, stream>>>(
      xf1, lin1W, lin1b, lin2W, lin2b, (float*)d_out, B, N);
}
